// Attention_81475529605197
// MI455X (gfx1250) — matrix operand order repaired and checked by reading, not compiled, not run on hardware
//
#include <hip/hip_runtime.h>
#include <hip/hip_bf16.h>

// B=8, T=2048, C=1024, HD=64. Flash-style causal attention with bf16 WMMA.

typedef __attribute__((ext_vector_type(16))) __bf16        v16bf;
typedef __attribute__((ext_vector_type(8)))  float         v8f;
typedef __attribute__((ext_vector_type(4)))  unsigned int  v4u;

union BFrag {
    v16bf bf;
    v4u   u4[2];
};

__device__ __forceinline__ unsigned short f2bf(float f) {
    unsigned int u = __float_as_uint(f);
    unsigned int r = 0x7FFFu + ((u >> 16) & 1u);   // round to nearest even
    return (unsigned short)((u + r) >> 16);
}

// ---------------------------------------------------------------------------
// Pass 0: convert Wq/Wk/Wv [C=1024, HD=64] fp32 -> bf16 transposed [64, 1024]
// so that WMMA B-fragments (K-major per lane) are contiguous loads.
// ---------------------------------------------------------------------------
__global__ void wcvt_kernel(const float* __restrict__ Wq,
                            const float* __restrict__ Wk,
                            const float* __restrict__ Wv,
                            unsigned short* __restrict__ wqt,
                            unsigned short* __restrict__ wkt,
                            unsigned short* __restrict__ wvt) {
    int tid = blockIdx.x * blockDim.x + threadIdx.x;
    if (tid >= 3 * 64 * 1024) return;
    int m   = tid >> 16;       // which matrix
    int rem = tid & 65535;
    int n   = rem >> 10;       // 0..63  (HD)
    int k   = rem & 1023;      // 0..1023 (C)
    const float* W = (m == 0) ? Wq : (m == 1) ? Wk : Wv;
    unsigned short* WT = (m == 0) ? wqt : (m == 1) ? wkt : wvt;
    WT[n * 1024 + k] = f2bf(W[k * 64 + n]);
}

// ---------------------------------------------------------------------------
// Pass 1: q = x@Wq, k = x@Wk, v = x@Wv  (M = B*T = 16384, K = 1024, N = 64 x3)
// One block = one 16-row M-tile. x tile staged as bf16 in LDS (32 KB).
// 4 waves; wave w owns output-column tile w (16 cols) of q, k and v.
// q,k stored [B*T, 64] bf16; v stored transposed [B, 64, T] bf16.
// ---------------------------------------------------------------------------
__global__ __launch_bounds__(128) void qkv_kernel(
    const float* __restrict__ x,
    const unsigned short* __restrict__ wqt,
    const unsigned short* __restrict__ wkt,
    const unsigned short* __restrict__ wvt,
    unsigned short* __restrict__ qb,
    unsigned short* __restrict__ kb,
    unsigned short* __restrict__ vtb)
{
    __shared__ __align__(16) unsigned short xlds[16 * 1024];
    const int tid = threadIdx.x;
    const int gm0 = blockIdx.x * 16;

    // cooperative stage: 16 x 1024 fp32 -> bf16 LDS, float4 loads
    for (int i = 0; i < 32; ++i) {
        int f    = tid + i * 128;          // float4 index 0..4095
        int row  = f >> 8;
        int colv = f & 255;
        const float4 v = *(const float4*)(x + (gm0 + row) * 1024 + colv * 4);
        unsigned int p0 = (unsigned int)f2bf(v.x) | ((unsigned int)f2bf(v.y) << 16);
        unsigned int p1 = (unsigned int)f2bf(v.z) | ((unsigned int)f2bf(v.w) << 16);
        *(uint2*)(&xlds[row * 1024 + colv * 4]) = make_uint2(p0, p1);
    }
    __syncthreads();

    const int lane = tid & 31;
    const int wave = tid >> 5;            // 0..3 -> n-col tile
    const int half = lane >> 4;
    const int l16  = lane & 15;

    v8f cq = {}, ck = {}, cv = {};

    // A layout (16x32 bf16): lane m<16 -> row m, K 0..15; lane m+16 -> K 16..31
    const int arow  = l16;
    const int abase = half * 16;
    // B layout (32x16 bf16): lane n<16 -> col n, K 0..15; lane n+16 -> K 16..31
    const int nrow   = wave * 16 + l16;
    const int kobase = half * 16;

    for (int k0 = 0; k0 < 1024; k0 += 32) {
        BFrag a;
        a.u4[0] = *(const v4u*)(&xlds[arow * 1024 + k0 + abase]);
        a.u4[1] = *(const v4u*)(&xlds[arow * 1024 + k0 + abase + 8]);

        const unsigned short* pq = wqt + nrow * 1024 + k0 + kobase;
        const unsigned short* pk = wkt + nrow * 1024 + k0 + kobase;
        const unsigned short* pv = wvt + nrow * 1024 + k0 + kobase;
        BFrag bq, bk2, bv2;
        bq.u4[0]  = *(const v4u*)(pq);     bq.u4[1]  = *(const v4u*)(pq + 8);
        bk2.u4[0] = *(const v4u*)(pk);     bk2.u4[1] = *(const v4u*)(pk + 8);
        bv2.u4[0] = *(const v4u*)(pv);     bv2.u4[1] = *(const v4u*)(pv + 8);

        cq = __builtin_amdgcn_wmma_f32_16x16x32_bf16(false, a.bf, false, bq.bf,  (short)0, cq, false, false);
        ck = __builtin_amdgcn_wmma_f32_16x16x32_bf16(false, a.bf, false, bk2.bf, (short)0, ck, false, false);
        cv = __builtin_amdgcn_wmma_f32_16x16x32_bf16(false, a.bf, false, bv2.bf, (short)0, cv, false, false);
    }

    // C layout: VGPR r -> row r (lanes 0-15) / row r+8 (lanes 16-31), col = lane&15
    for (int r = 0; r < 8; ++r) {
        int gm  = gm0 + r + 8 * half;
        int col = wave * 16 + l16;
        qb[gm * 64 + col] = f2bf(cq[r]);
        kb[gm * 64 + col] = f2bf(ck[r]);
        int b = gm >> 11;          // / 2048
        int t = gm & 2047;
        vtb[(b * 64 + col) * 2048 + t] = f2bf(cv[r]);
    }
}

// ---------------------------------------------------------------------------
// Pass 2: flash attention. One wave per 16-query tile, 32 keys per iteration:
//   S(16x32) = Q(16x64) Kt  -> online softmax (shfl reductions per 16-lane
//   half = one C-layout row) -> P relayout via per-wave LDS -> O += P V.
// ---------------------------------------------------------------------------
__global__ __launch_bounds__(128) void attn_kernel(
    const unsigned short* __restrict__ qb,
    const unsigned short* __restrict__ kb,
    const unsigned short* __restrict__ vtb,
    float* __restrict__ out)
{
    __shared__ __align__(16) unsigned short plds[4][16 * 32];
    const int tid  = threadIdx.x;
    const int lane = tid & 31;
    const int wave = tid >> 5;
    const int half = lane >> 4;
    const int l16  = lane & 15;
    const int b    = blockIdx.y;
    const int qt   = blockIdx.x * 4 + wave;       // 0..127

    // Q A-fragments for K-dim steps d0 = 0 and 32 of HD=64
    BFrag aq[2];
    {
        const unsigned short* qp = qb + (b * 2048 + qt * 16 + l16) * 64;
        for (int d = 0; d < 2; ++d) {
            int base = d * 32 + half * 16;
            aq[d].u4[0] = *(const v4u*)(qp + base);
            aq[d].u4[1] = *(const v4u*)(qp + base + 8);
        }
    }

    v8f co[4] = {};
    float mrow[8], lrow[8], alpha[8];
    for (int r = 0; r < 8; ++r) { mrow[r] = -3.0e38f; lrow[r] = 0.0f; }

    const float scale = 0.03125f;                  // C^-0.5 = 1/32
    const int   jmax  = (qt + 1) * 16;
    unsigned short* pl = plds[wave];

    for (int j0 = 0; j0 < jmax; j0 += 32) {
        // S = Q Kt : two 16-key N-tiles, K-dim 64 (2 wmma steps each)
        v8f cs[2] = {};
        for (int t = 0; t < 2; ++t) {
            const unsigned short* kp =
                kb + (b * 2048 + j0 + 16 * t + l16) * 64 + half * 16;
            for (int d = 0; d < 2; ++d) {
                BFrag bk2;
                bk2.u4[0] = *(const v4u*)(kp + d * 32);
                bk2.u4[1] = *(const v4u*)(kp + d * 32 + 8);
                cs[t] = __builtin_amdgcn_wmma_f32_16x16x32_bf16(
                    false, aq[d].bf, false, bk2.bf, (short)0, cs[t], false, false);
            }
        }

        // online softmax per row (row = 16-lane half of one accumulator VGPR)
        for (int r = 0; r < 8; ++r) {
            int qrow = qt * 16 + r + 8 * half;
            float s0 = cs[0][r] * scale;
            float s1 = cs[1][r] * scale;
            if (j0 + l16      > qrow) s0 = -3.0e38f;   // causal mask
            if (j0 + 16 + l16 > qrow) s1 = -3.0e38f;
            float rm = fmaxf(s0, s1);
            for (int m = 1; m < 16; m <<= 1)
                rm = fmaxf(rm, __shfl_xor(rm, m, 32));
            float mnew = fmaxf(mrow[r], rm);
            float a    = __expf(mrow[r] - mnew);
            float p0   = __expf(s0 - mnew);
            float p1   = __expf(s1 - mnew);
            pl[(r + 8 * half) * 32 + l16]      = f2bf(p0);
            pl[(r + 8 * half) * 32 + 16 + l16] = f2bf(p1);
            float rs = p0 + p1;
            for (int m = 1; m < 16; m <<= 1)
                rs += __shfl_xor(rs, m, 32);
            lrow[r]  = lrow[r] * a + rs;
            mrow[r]  = mnew;
            alpha[r] = a;
        }
        for (int t = 0; t < 4; ++t)
            for (int r = 0; r < 8; ++r)
                co[t][r] *= alpha[r];

        // cross-lane DS RAW: all lanes' P stores must land before A-frag reads
        asm volatile("s_wait_dscnt 0" ::: "memory");

        BFrag ap;
        ap.u4[0] = *(const v4u*)(&pl[l16 * 32 + half * 16]);
        ap.u4[1] = *(const v4u*)(&pl[l16 * 32 + 8 + half * 16]);

        // O += P V : V B-fragments contiguous thanks to [B, HD, T] layout
        for (int t = 0; t < 4; ++t) {
            const unsigned short* vp =
                vtb + (b * 64 + t * 16 + l16) * 2048 + j0 + half * 16;
            BFrag bv2;
            bv2.u4[0] = *(const v4u*)(vp);
            bv2.u4[1] = *(const v4u*)(vp + 8);
            co[t] = __builtin_amdgcn_wmma_f32_16x16x32_bf16(
                false, ap.bf, false, bv2.bf, (short)0, co[t], false, false);
        }
    }

    for (int r = 0; r < 8; ++r) {
        int trow  = qt * 16 + r + 8 * half;
        float inv = 1.0f / lrow[r];
        for (int t = 0; t < 4; ++t)
            out[(b * 2048 + trow) * 64 + t * 16 + l16] = co[t][r] * inv;
    }
}

// ---------------------------------------------------------------------------
extern "C" void kernel_launch(void* const* d_in, const int* in_sizes, int n_in,
                              void* d_out, int out_size, void* d_ws, size_t ws_size,
                              hipStream_t stream) {
    (void)in_sizes; (void)n_in; (void)out_size; (void)ws_size;
    const float* x  = (const float*)d_in[0];
    const float* Wq = (const float*)d_in[1];
    const float* Wk = (const float*)d_in[2];
    const float* Wv = (const float*)d_in[3];
    float* out = (float*)d_out;

    unsigned short* ws  = (unsigned short*)d_ws;
    unsigned short* wqt = ws;                       // [64,1024] bf16
    unsigned short* wkt = ws + 64 * 1024;
    unsigned short* wvt = ws + 2 * 64 * 1024;
    unsigned short* qb  = ws + 3 * 64 * 1024;       // [B*T,64] bf16
    unsigned short* kb  = qb + 16384 * 64;          // [B*T,64] bf16
    unsigned short* vtb = kb + 16384 * 64;          // [B,64,T] bf16
    // total workspace: ~6.7 MB

    wcvt_kernel<<<768, 256, 0, stream>>>(Wq, Wk, Wv, wqt, wkt, wvt);
    qkv_kernel<<<1024, 128, 0, stream>>>(x, wqt, wkt, wvt, qb, kb, vtb);
    attn_kernel<<<dim3(32, 8), 128, 0, stream>>>(qb, kb, vtb, out);
}
